// Informer_74921409511840
// MI455X (gfx1250) — matrix operand order repaired and checked by reading, not compiled, not run on hardware
//
#include <hip/hip_runtime.h>
#include <hip/hip_bf16.h>
#include <math.h>

// ---------------------------------------------------------------------------
// Model constants (Informer)
// ---------------------------------------------------------------------------
constexpr int kB = 8, kLS = 1024, kLD = 512, kCIN = 16, kD = 512, kDFF = 512;
constexpr int kH = 8, kE = 64, kU = 35, kPred = 256;

typedef __attribute__((ext_vector_type(16))) __bf16 v16bf;
typedef __attribute__((ext_vector_type(8)))  __bf16 v8bf;
typedef __attribute__((ext_vector_type(8)))  float  v8f;

// ---------------------------------------------------------------------------
// fp32 -> bf16 elementwise (one-time weight conversion into workspace)
// ---------------------------------------------------------------------------
__global__ void f32_to_bf16_kernel(const float* __restrict__ x, __bf16* __restrict__ y, int n) {
  int i = blockIdx.x * blockDim.x + threadIdx.x;
  if (i < n) y[i] = (__bf16)x[i];
}

// ---------------------------------------------------------------------------
// WMMA NT-GEMM:  C[M,N] = act(A[M,K] * W[N,K]^T + bias[N])
//   A: fp32 (converted to bf16 in-register), W: pre-converted bf16
//   One wave computes a 32x64 tile: 2 M-subtiles x 4 N-subtiles,
//   8x v_wmma_f32_16x16x32_bf16 per K-step of 32; B fragments reused 2x.
//   out_mode 0: row-major C (M x N)
//   out_mode 1: scatter to (B,H,Lseq,E) layout (QKV projections)
// ---------------------------------------------------------------------------
__device__ __forceinline__ float gelu_exact(float x) {
  return 0.5f * x * (1.0f + erff(x * 0.7071067811865476f));
}

__device__ __forceinline__ void pack8(v16bf& d, int base, float4 a, float4 b) {
  d[base + 0] = (__bf16)a.x; d[base + 1] = (__bf16)a.y;
  d[base + 2] = (__bf16)a.z; d[base + 3] = (__bf16)a.w;
  d[base + 4] = (__bf16)b.x; d[base + 5] = (__bf16)b.y;
  d[base + 6] = (__bf16)b.z; d[base + 7] = (__bf16)b.w;
}

__global__ void gemm_nt_wmma(const float* __restrict__ A, const __bf16* __restrict__ W,
                             const float* __restrict__ bias, float* __restrict__ C,
                             int M, int N, int K, int act, int out_mode, int Lseq) {
  const int mt    = blockIdx.x * 32;
  const int nt    = blockIdx.y * 64;
  const int lane  = threadIdx.x;        // 0..31
  const int mrow  = lane & 15;          // A: row within 16-row subtile
  const int khalf = lane >> 4;          // A: lo/hi K-half of lane group
  const int ncol  = lane & 15;          // B: column within subtile (= W row)
  const int klo   = khalf * 16;         // B: K-half per lane group

  v8f acc[2][4] = {};
  const float* A0 = A + (size_t)(mt + mrow) * K + khalf * 16;
  const float* A1 = A0 + (size_t)16 * K;
  const __bf16* Wb = W + (size_t)(nt + ncol) * K + klo;

  for (int k0 = 0; k0 < K; k0 += 32) {
    // A fragments: contiguous k order (same as B) — lane m holds K = khalf*16 .. +15
    v16bf af0, af1;
    {
      const float4* p = (const float4*)(A0 + k0);
      const float4* q = (const float4*)(A0 + k0 + 8);
      pack8(af0, 0, p[0], p[1]);
      pack8(af0, 8, q[0], q[1]);
      p = (const float4*)(A1 + k0);
      q = (const float4*)(A1 + k0 + 8);
      pack8(af1, 0, p[0], p[1]);
      pack8(af1, 8, q[0], q[1]);
    }
#pragma unroll
    for (int t = 0; t < 4; ++t) {
      // B fragment: column n = W row (nt + t*16 + ncol), K = k0+klo .. +15 contiguous bf16
      const v8bf* pw = (const v8bf*)(Wb + (size_t)t * 16 * K + k0);
      v8bf lo = pw[0], hi = pw[1];
      v16bf bf = __builtin_shufflevector(lo, hi, 0, 1, 2, 3, 4, 5, 6, 7,
                                         8, 9, 10, 11, 12, 13, 14, 15);
      acc[0][t] = __builtin_amdgcn_wmma_f32_16x16x32_bf16(false, af0, false, bf,
                                                          (short)0, acc[0][t], false, false);
      acc[1][t] = __builtin_amdgcn_wmma_f32_16x16x32_bf16(false, af1, false, bf,
                                                          (short)0, acc[1][t], false, false);
    }
  }

#pragma unroll
  for (int mi = 0; mi < 2; ++mi) {
#pragma unroll
    for (int t = 0; t < 4; ++t) {
      int col = nt + t * 16 + ncol;
      float bb = bias ? bias[col] : 0.0f;
#pragma unroll
      for (int r = 0; r < 8; ++r) {
        int row = mt + mi * 16 + khalf * 8 + r;   // C layout: lane half selects M 0-7 / 8-15
        float v = acc[mi][t][r] + bb;
        if (act == 1) v = gelu_exact(v);
        if (out_mode == 0) {
          C[(size_t)row * N + col] = v;
        } else {
          int b = row / Lseq, l = row - b * Lseq;
          int h = col >> 6, e = col & 63;
          C[(((size_t)(b * kH + h) * Lseq) + l) * kE + e] = v;
        }
      }
    }
  }
}

// ---------------------------------------------------------------------------
// Token conv embedding (circular pad, k=3, CIN->D) + sinusoidal pos embed
// ---------------------------------------------------------------------------
__global__ void token_conv_embed(const float* __restrict__ x, const float* __restrict__ W,
                                 const float* __restrict__ bias, float* __restrict__ y, int L) {
  int idx = blockIdx.x * blockDim.x + threadIdx.x;   // over B*L*D
  int d = idx % kD;
  int l = (idx / kD) % L;
  int b = idx / (kD * L);
  float acc = bias[d];
#pragma unroll
  for (int t = 0; t < 3; ++t) {
    int ls = l + t - 1;
    if (ls < 0) ls += L;
    if (ls >= L) ls -= L;
    const float* xr = x + ((size_t)b * L + ls) * kCIN;
    const float* wr = W + (size_t)d * kCIN * 3 + t;
#pragma unroll
    for (int c = 0; c < kCIN; ++c) acc += xr[c] * wr[c * 3];
  }
  int i2 = d & ~1;
  float div = __expf(-(float)i2 * (logf(10000.0f) / (float)kD));
  float ang = (float)l * div;
  acc += (d & 1) ? __cosf(ang) : __sinf(ang);
  y[idx] = acc;
}

// ---------------------------------------------------------------------------
// Depthwise distil conv, k=3, zero pad (1,1)
// ---------------------------------------------------------------------------
__global__ void distil_conv_kernel(const float* __restrict__ x, const float* __restrict__ W,
                                   const float* __restrict__ bias, float* __restrict__ y, int L) {
  int idx = blockIdx.x * blockDim.x + threadIdx.x;   // over B*L*D
  int d = idx % kD;
  int l = (idx / kD) % L;
  int b = idx / (kD * L);
  float acc = bias[d];
#pragma unroll
  for (int t = 0; t < 3; ++t) {
    int ls = l + t - 1;
    if (ls >= 0 && ls < L) acc += x[((size_t)b * L + ls) * kD + d] * W[d * 3 + t];
  }
  y[idx] = acc;
}

// ---------------------------------------------------------------------------
// LayerNorm over D=512: out = LN(a + badd) * g + beta  (badd optional)
// block = 128 threads, one row per block
// ---------------------------------------------------------------------------
__global__ void layernorm_kernel(const float* __restrict__ a, const float* __restrict__ badd,
                                 const float* __restrict__ g, const float* __restrict__ beta,
                                 float* __restrict__ out) {
  int row = blockIdx.x;
  int tid = threadIdx.x;
  __shared__ float red[128];
  __shared__ float s_mean, s_rstd;
  const float* ar = a + (size_t)row * kD;
  const float* br = badd ? badd + (size_t)row * kD : nullptr;
  float vals[4];
  float sum = 0.0f;
#pragma unroll
  for (int i = 0; i < 4; ++i) {
    int c = tid + i * 128;
    float v = ar[c] + (br ? br[c] : 0.0f);
    vals[i] = v;
    sum += v;
  }
  red[tid] = sum; __syncthreads();
  for (int s = 64; s > 0; s >>= 1) { if (tid < s) red[tid] += red[tid + s]; __syncthreads(); }
  if (tid == 0) s_mean = red[0] / (float)kD;
  __syncthreads();
  float m = s_mean;
  float vs = 0.0f;
#pragma unroll
  for (int i = 0; i < 4; ++i) { float dd = vals[i] - m; vs += dd * dd; }
  red[tid] = vs; __syncthreads();
  for (int s = 64; s > 0; s >>= 1) { if (tid < s) red[tid] += red[tid + s]; __syncthreads(); }
  if (tid == 0) s_rstd = rsqrtf(red[0] / (float)kD + 1e-5f);
  __syncthreads();
  float r = s_rstd;
#pragma unroll
  for (int i = 0; i < 4; ++i) {
    int c = tid + i * 128;
    out[(size_t)row * kD + c] = (vals[i] - m) * r * g[c] + beta[c];
  }
}

// ---------------------------------------------------------------------------
// ProbSparse attention pieces. Q/K/V stored as (B,H,L,E).
// ---------------------------------------------------------------------------
__global__ void prob_m_kernel(const float* __restrict__ Qb, const float* __restrict__ Kb,
                              float* __restrict__ M, int Lq, int Lk) {
  int wid  = (blockIdx.x * blockDim.x + threadIdx.x) >> 5;
  int lane = threadIdx.x & 31;
  int l  = wid % Lq;
  int bh = wid / Lq;
  const float* q = Qb + ((size_t)bh * Lq + l) * kE;
  float q0 = q[lane], q1 = q[lane + 32];
  float mx = -1e30f, sm = 0.0f;
  for (int j = 0; j < kU; ++j) {
    unsigned h = ((unsigned)wid * 2654435761u) ^ ((unsigned)(j + 1) * 2246822519u);
    h ^= h >> 15; h *= 2654435761u; h ^= h >> 13;
    int kidx = (int)(h % (unsigned)Lk);
    const float* kr = Kb + ((size_t)bh * Lk + kidx) * kE;
    float p = q0 * kr[lane] + q1 * kr[lane + 32];
    for (int off = 16; off > 0; off >>= 1) p += __shfl_xor(p, off, 32);
    mx = fmaxf(mx, p);
    sm += p;
  }
  if (lane == 0) M[wid] = mx - sm / (float)Lk;
}

__global__ void topk_kernel(float* __restrict__ M, int* __restrict__ top, int Lq) {
  int bh  = blockIdx.x;
  int tid = threadIdx.x;   // 256
  __shared__ float sv[256];
  __shared__ int   si[256];
  float* m = M + (size_t)bh * Lq;
  for (int it = 0; it < kU; ++it) {
    float best = -1e30f; int bi = 0;
    for (int c = tid; c < Lq; c += 256) {
      float v = m[c];
      if (v > best) { best = v; bi = c; }
    }
    sv[tid] = best; si[tid] = bi; __syncthreads();
    for (int s = 128; s > 0; s >>= 1) {
      if (tid < s && sv[tid + s] > sv[tid]) { sv[tid] = sv[tid + s]; si[tid] = si[tid + s]; }
      __syncthreads();
    }
    if (tid == 0) { top[bh * kU + it] = si[0]; m[si[0]] = -1e30f; }
    __syncthreads();
  }
}

__global__ void vmean_kernel(const float* __restrict__ Vb, float* __restrict__ vm, int Lk) {
  int bh = blockIdx.x;
  int e  = threadIdx.x;   // 64
  const float* v = Vb + (size_t)bh * Lk * kE + e;
  float s = 0.0f;
  for (int l = 0; l < Lk; ++l) s += v[(size_t)l * kE];
  vm[bh * kE + e] = s / (float)Lk;
}

__global__ void ctx_fill_kernel(const float* __restrict__ vm, float* __restrict__ ctx, int Lq) {
  int idx = blockIdx.x * blockDim.x + threadIdx.x;   // B*H*Lq*E
  int e  = idx % kE;
  int bh = idx / (Lq * kE);
  ctx[idx] = vm[bh * kE + e];
}

__global__ void attn_update_kernel(const float* __restrict__ Qb, const float* __restrict__ Kb,
                                   const float* __restrict__ Vb, const int* __restrict__ top,
                                   float* __restrict__ ctx, int Lq, int Lk) {
  __shared__ float sc[1024];           // Lk <= 1024
  __shared__ float qs[kE];
  __shared__ float red[256];
  __shared__ float outp[4][kE];
  int bh  = blockIdx.x;
  int ui  = blockIdx.y;
  int tid = threadIdx.x;               // 256
  int ql  = top[bh * kU + ui];
  if (tid < kE) qs[tid] = Qb[((size_t)bh * Lq + ql) * kE + tid];
  __syncthreads();
  float lmax = -1e30f;
  for (int kk = tid; kk < Lk; kk += 256) {
    const float* kr = Kb + ((size_t)bh * Lk + kk) * kE;
    float dot = 0.0f;
#pragma unroll 8
    for (int e = 0; e < kE; ++e) dot += qs[e] * kr[e];
    dot *= 0.125f;                     // 1/sqrt(E)
    sc[kk] = dot;
    lmax = fmaxf(lmax, dot);
  }
  red[tid] = lmax; __syncthreads();
  for (int s = 128; s > 0; s >>= 1) { if (tid < s) red[tid] = fmaxf(red[tid], red[tid + s]); __syncthreads(); }
  float gmax = red[0]; __syncthreads();
  float lsum = 0.0f;
  for (int kk = tid; kk < Lk; kk += 256) { float ev = __expf(sc[kk] - gmax); sc[kk] = ev; lsum += ev; }
  red[tid] = lsum; __syncthreads();
  for (int s = 128; s > 0; s >>= 1) { if (tid < s) red[tid] += red[tid + s]; __syncthreads(); }
  float inv = 1.0f / red[0]; __syncthreads();
  int e = tid & 63, chunk = tid >> 6;
  float acc = 0.0f;
  for (int kk = chunk; kk < Lk; kk += 4) acc += sc[kk] * Vb[((size_t)bh * Lk + kk) * kE + e];
  outp[chunk][e] = acc; __syncthreads();
  if (tid < kE) {
    float o = (outp[0][tid] + outp[1][tid] + outp[2][tid] + outp[3][tid]) * inv;
    ctx[((size_t)bh * Lq + ql) * kE + tid] = o;
  }
}

// (B,H,L,E) -> (B,L,H*E) transpose (mix=False path)
__global__ void bhle_to_bld_kernel(const float* __restrict__ ctx, float* __restrict__ out, int L) {
  int idx = blockIdx.x * blockDim.x + threadIdx.x;   // B*L*D
  int e = idx % kE;
  int h = (idx / kE) % kH;
  int l = (idx / kD) % L;
  int b = idx / (kD * L);
  out[idx] = ctx[(((size_t)b * kH + h) * L + l) * kE + e];
}

// Final projection D -> 1 over last kPred positions
__global__ void proj_kernel(const float* __restrict__ dec, const float* __restrict__ pW,
                            const float* __restrict__ pb, float* __restrict__ out) {
  int o = blockIdx.x;                  // B*kPred
  int lane = threadIdx.x;              // 32
  int b = o / kPred;
  int i = o % kPred;
  const float* row = dec + ((size_t)b * kLD + (kLD - kPred) + i) * kD;
  float s = 0.0f;
  for (int c = lane; c < kD; c += 32) s += row[c] * pW[c];
  for (int off = 16; off > 0; off >>= 1) s += __shfl_xor(s, off, 32);
  if (lane == 0) out[o] = s + pb[0];
}

// ---------------------------------------------------------------------------
// Host orchestration
// ---------------------------------------------------------------------------
extern "C" void kernel_launch(void* const* d_in, const int* in_sizes, int n_in,
                              void* d_out, int out_size, void* d_ws, size_t ws_size,
                              hipStream_t stream) {
  (void)in_sizes; (void)n_in; (void)out_size; (void)ws_size;
  const float* x_enc      = (const float*)d_in[0];
  const float* x_dec      = (const float*)d_in[1];
  const float* enc_emb_W  = (const float*)d_in[2];
  const float* enc_emb_b  = (const float*)d_in[3];
  const float* dec_emb_W  = (const float*)d_in[4];
  const float* dec_emb_b  = (const float*)d_in[5];
  const float* enc_attn_W = (const float*)d_in[6];
  const float* enc_attn_b = (const float*)d_in[7];
  const float* enc_ff_W1  = (const float*)d_in[8];
  const float* enc_ff_b1  = (const float*)d_in[9];
  const float* enc_ff_W2  = (const float*)d_in[10];
  const float* enc_ff_b2  = (const float*)d_in[11];
  const float* enc_ln_g   = (const float*)d_in[12];
  const float* enc_ln_b   = (const float*)d_in[13];
  const float* distil_W   = (const float*)d_in[14];
  const float* distil_b   = (const float*)d_in[15];
  const float* enc_norm_g = (const float*)d_in[16];
  const float* enc_norm_b = (const float*)d_in[17];
  const float* dec_self_W = (const float*)d_in[18];
  const float* dec_self_b = (const float*)d_in[19];
  const float* dec_cross_W= (const float*)d_in[20];
  const float* dec_cross_b= (const float*)d_in[21];
  const float* dec_ff_W1  = (const float*)d_in[22];
  const float* dec_ff_b1  = (const float*)d_in[23];
  const float* dec_ff_W2  = (const float*)d_in[24];
  const float* dec_ff_b2  = (const float*)d_in[25];
  const float* dec_ln_g   = (const float*)d_in[26];
  const float* dec_ln_b   = (const float*)d_in[27];
  const float* dec_norm_g = (const float*)d_in[28];
  const float* dec_norm_b = (const float*)d_in[29];
  const float* proj_W     = (const float*)d_in[30];
  const float* proj_b     = (const float*)d_in[31];
  float* out = (float*)d_out;

  // Byte-based workspace bump allocator (~150 MB total; L2 is 192 MB)
  char* ws = (char*)d_ws;
  size_t off = 0;
  auto balloc = [&](size_t bytes) {
    void* p = ws + off;
    off += (bytes + 255) & ~(size_t)255;
    return p;
  };
  const size_t SL = (size_t)kB * kLS * kD;            // 4M elements
  float* S_ENC  = (float*)balloc(SL * 4);
  float* S_Q    = (float*)balloc(SL * 4);
  float* S_K    = (float*)balloc(SL * 4);
  float* S_V    = (float*)balloc(SL * 4);
  float* S_CTX  = (float*)balloc(SL * 4);
  float* S_T1   = (float*)balloc(SL * 4);
  float* S_T2   = (float*)balloc(SL * 4);
  float* S_DECA = (float*)balloc((size_t)kB * kLD * kD * 4);
  float* S_DECB = (float*)balloc((size_t)kB * kLD * kD * 4);
  float* S_M    = (float*)balloc((size_t)kB * kH * kLS * 4);
  float* S_VM   = (float*)balloc((size_t)kB * kH * kE * 4);
  int*   S_TOP  = (int*)balloc((size_t)kB * kH * kU * 4);

  // One-time fp32 -> bf16 weight conversion (weights then load at 2B/elem and
  // the GEMM hot loop has no W-side converts).
  auto cvtw = [&](const float* src, size_t n) {
    __bf16* dst = (__bf16*)balloc(n * sizeof(__bf16));
    f32_to_bf16_kernel<<<(int)((n + 255) / 256), 256, 0, stream>>>(src, dst, (int)n);
    return dst;
  };
  const size_t DD = (size_t)kD * kD;
  __bf16* Wb_enc_attn = cvtw(enc_attn_W, 3 * 4 * DD);
  __bf16* Wb_enc_ff1  = cvtw(enc_ff_W1, 3 * (size_t)kDFF * kD);
  __bf16* Wb_enc_ff2  = cvtw(enc_ff_W2, 3 * (size_t)kD * kDFF);
  __bf16* Wb_dec_self = cvtw(dec_self_W, 2 * 4 * DD);
  __bf16* Wb_dec_crs  = cvtw(dec_cross_W, 2 * 4 * DD);
  __bf16* Wb_dec_ff1  = cvtw(dec_ff_W1, 2 * (size_t)kDFF * kD);
  __bf16* Wb_dec_ff2  = cvtw(dec_ff_W2, 2 * (size_t)kD * kDFF);

  auto ln = [&](const float* a, const float* badd, const float* g, const float* be,
                float* o, int rows) {
    layernorm_kernel<<<rows, 128, 0, stream>>>(a, badd, g, be, o);
  };

  auto gemm = [&](const float* A, const __bf16* W, const float* bias, float* C,
                  int M, int N, int K, int act, int out_mode, int Lseq) {
    gemm_nt_wmma<<<dim3(M / 32, N / 64), 32, 0, stream>>>(A, W, bias, C, M, N, K, act, out_mode, Lseq);
  };

  // ProbSparse attention block: writes (B*Lq, D) attention output to outBuf.
  auto attention = [&](const float* xq, const float* xkv, const __bf16* Wb, const float* bb,
                       int Lq, int Lk, int mix, float* outBuf) {
    const __bf16* Wq = Wb;           const float* bq = bb;
    const __bf16* Wk = Wb + DD;      const float* bk = bb + kD;
    const __bf16* Wv = Wb + 2 * DD;  const float* bv = bb + 2 * kD;
    const __bf16* Wo = Wb + 3 * DD;  const float* bo = bb + 3 * kD;
    gemm(xq,  Wq, bq, S_Q, kB * Lq, kD, kD, 0, 1, Lq);
    gemm(xkv, Wk, bk, S_K, kB * Lk, kD, kD, 0, 1, Lk);
    gemm(xkv, Wv, bv, S_V, kB * Lk, kD, kD, 0, 1, Lk);
    prob_m_kernel<<<(kB * kH * Lq) / 8, 256, 0, stream>>>(S_Q, S_K, S_M, Lq, Lk);
    topk_kernel<<<kB * kH, 256, 0, stream>>>(S_M, S_TOP, Lq);
    vmean_kernel<<<kB * kH, kE, 0, stream>>>(S_V, S_VM, Lk);
    ctx_fill_kernel<<<(kB * kH * Lq * kE) / 256, 256, 0, stream>>>(S_VM, S_CTX, Lq);
    attn_update_kernel<<<dim3(kB * kH, kU), 256, 0, stream>>>(S_Q, S_K, S_V, S_TOP, S_CTX, Lq, Lk);
    const float* oin = S_CTX;           // mix=True: (B,H,L,E) flat == (B*L, D) reinterpret
    if (!mix) {
      bhle_to_bld_kernel<<<(kB * Lq * kD) / 256, 256, 0, stream>>>(S_CTX, S_T1, Lq);
      oin = S_T1;
    }
    gemm(oin, Wo, bo, outBuf, kB * Lq, kD, kD, 0, 0, 0);
  };

  // ----------------- Encoder -----------------
  token_conv_embed<<<(kB * kLS * kD) / 256, 256, 0, stream>>>(x_enc, enc_emb_W, enc_emb_b, S_ENC, kLS);
  for (int i = 0; i < 3; ++i) {
    attention(S_ENC, S_ENC, Wb_enc_attn + (size_t)i * 4 * DD,
              enc_attn_b + (size_t)i * 4 * kD, kLS, kLS, 0, S_T2);
    ln(S_ENC, S_T2, enc_ln_g + (i * 2 + 0) * kD, enc_ln_b + (i * 2 + 0) * kD, S_T1, kB * kLS);
    gemm(S_T1, Wb_enc_ff1 + (size_t)i * kDFF * kD, enc_ff_b1 + i * kDFF, S_T2,
         kB * kLS, kDFF, kD, 1, 0, 0);                       // GELU
    gemm(S_T2, Wb_enc_ff2 + (size_t)i * kD * kDFF, enc_ff_b2 + i * kD, S_Q,
         kB * kLS, kD, kDFF, 0, 0, 0);
    ln(S_T1, S_Q, enc_ln_g + (i * 2 + 1) * kD, enc_ln_b + (i * 2 + 1) * kD, S_CTX, kB * kLS);
    if (i < 2) {
      distil_conv_kernel<<<(kB * kLS * kD) / 256, 256, 0, stream>>>(
          S_CTX, distil_W + (size_t)i * kD * 3, distil_b + i * kD, S_ENC, kLS);
    } else {
      ln(S_CTX, nullptr, enc_norm_g, enc_norm_b, S_ENC, kB * kLS);
    }
  }

  // ----------------- Decoder -----------------
  token_conv_embed<<<(kB * kLD * kD) / 256, 256, 0, stream>>>(x_dec, dec_emb_W, dec_emb_b, S_DECA, kLD);
  float* dec  = S_DECA;
  float* dalt = S_DECB;
  for (int i = 0; i < 2; ++i) {
    attention(dec, dec, Wb_dec_self + (size_t)i * 4 * DD,
              dec_self_b + (size_t)i * 4 * kD, kLD, kLD, 1, S_T2);          // mix=True
    ln(dec, S_T2, dec_ln_g + (i * 3 + 0) * kD, dec_ln_b + (i * 3 + 0) * kD, dalt, kB * kLD);
    { float* t = dec; dec = dalt; dalt = t; }
    attention(dec, S_ENC, Wb_dec_crs + (size_t)i * 4 * DD,
              dec_cross_b + (size_t)i * 4 * kD, kLD, kLS, 0, S_T2);         // cross, mix=False
    ln(dec, S_T2, dec_ln_g + (i * 3 + 1) * kD, dec_ln_b + (i * 3 + 1) * kD, S_T1, kB * kLD);
    gemm(S_T1, Wb_dec_ff1 + (size_t)i * kDFF * kD, dec_ff_b1 + i * kDFF, S_T2,
         kB * kLD, kDFF, kD, 1, 0, 0);                       // GELU
    gemm(S_T2, Wb_dec_ff2 + (size_t)i * kD * kDFF, dec_ff_b2 + i * kD, S_Q,
         kB * kLD, kD, kDFF, 0, 0, 0);
    ln(S_T1, S_Q, dec_ln_g + (i * 3 + 2) * kD, dec_ln_b + (i * 3 + 2) * kD, dalt, kB * kLD);
    { float* t = dec; dec = dalt; dalt = t; }
  }
  ln(dec, nullptr, dec_norm_g, dec_norm_b, S_T1, kB * kLD);
  proj_kernel<<<kB * kPred, 32, 0, stream>>>(S_T1, proj_W, proj_b, out);
}
